// Block_85194971284087
// MI455X (gfx1250) — matrix operand order repaired and checked by reading, not compiled, not run on hardware
//
#include <hip/hip_runtime.h>

// ---------------------------------------------------------------------------
// Types for CDNA5 WMMA (wave32): D(16x16,f32) = A(16x32,f16) x B(32x16,f16) + C
// ---------------------------------------------------------------------------
typedef _Float16 v16h __attribute__((ext_vector_type(16)));
typedef _Float16 h8   __attribute__((ext_vector_type(8)));
typedef float    v8f  __attribute__((ext_vector_type(8)));
typedef unsigned int u32x4 __attribute__((ext_vector_type(4)));
typedef int          i32x8 __attribute__((ext_vector_type(8)));
typedef int          i32x4 __attribute__((ext_vector_type(4)));

#ifndef __has_builtin
#define __has_builtin(x) 0
#endif
#if __has_builtin(__builtin_amdgcn_tensor_load_to_lds)
#define USE_TDM 1
#else
#define USE_TDM 0
#endif

union F16x16 { v16h v; h8 h[2]; };

__device__ inline v8f vzero8() {
    v8f z = {0.f,0.f,0.f,0.f,0.f,0.f,0.f,0.f};
    return z;
}

__device__ inline v8f wmma_f16(v16h a, v16h b, v8f c) {
    return __builtin_amdgcn_wmma_f32_16x16x32_f16(false, a, false, b, (short)0, c,
                                                  false, false);
}

// A fragment (16x32 f16): lane row = l&15; halves 0..15 -> K = 2*klo..2*klo+15
// contiguous, klo = (l>=16)?8:0. Two 16B loads.
__device__ inline v16h frag_a(const _Float16* row, int klo) {
    F16x16 u;
    u.h[0] = *(const h8*)(row + 2 * klo);
    u.h[1] = *(const h8*)(row + 2 * klo + 8);
    return u.v;
}

// B fragment (32x16 f16): lane col = l&15; K = khi..khi+15 contiguous,
// khi = (l>=16)?16:0. Pointer must already address (col, khi).
__device__ inline v16h frag_b(const _Float16* p) {
    F16x16 u;
    u.h[0] = *(const h8*)(p);
    u.h[1] = *(const h8*)(p + 8);
    return u.v;
}

__device__ inline float gelu_tanh(float x) {
    float u = 0.7978845608028654f * (x + 0.044715f * x * x * x);
    return 0.5f * x * (1.0f + tanhf(u));
}

#if USE_TDM
// TDM 2D tile load: tile_d0 x tile_d1 elements (2-byte), dim0 contiguous,
// row stride stride0 elements, into LDS at byte offset lds_off.
// D# group layout per CDNA5 ISA 8.3/8.4 (count=1, type=2 "image",
// data_size=1 -> 2 bytes, no gather/iterate/pad/multicast).
// clang-23 toolchain: 6-arg builtin (g0, g1, g2, g3, g_extra, cpol).
__device__ inline void tdm_load_tile_2d(unsigned lds_off, const _Float16* gaddr,
                                        unsigned tensor_d0, unsigned tensor_d1,
                                        unsigned tile_d0, unsigned tile_d1,
                                        unsigned stride0) {
    unsigned long long ga = (unsigned long long)(uintptr_t)gaddr;
    u32x4 g0;
    g0[0] = 1u;                                            // count=1 (valid)
    g0[1] = lds_off;                                       // lds_addr (bytes)
    g0[2] = (unsigned)ga;                                  // global_addr[31:0]
    g0[3] = (unsigned)((ga >> 32) & 0x1FFFFFFu) | (2u << 30);  // [56:32] | type=2
    i32x8 g1;
    g1[0] = 1 << 16;                                       // data_size=1 (2B)
    g1[1] = (int)((tensor_d0 & 0xFFFFu) << 16);            // tensor_dim0 lo
    g1[2] = (int)((tensor_d0 >> 16) | ((tensor_d1 & 0xFFFFu) << 16));
    g1[3] = (int)((tensor_d1 >> 16) | (tile_d0 << 16));    // tile_dim0
    g1[4] = (int)(tile_d1 & 0xFFFFu);                      // tile_dim1 (dim2=0)
    g1[5] = (int)stride0;                                  // tensor_dim0_stride lo
    g1[6] = 0;                                             // stride hi / dim1_stride lo
    g1[7] = 0;
    i32x4 gz4;
    gz4[0] = 0; gz4[1] = 0; gz4[2] = 0; gz4[3] = 0;
    i32x8 gz8;
    gz8[0] = 0; gz8[1] = 0; gz8[2] = 0; gz8[3] = 0;
    gz8[4] = 0; gz8[5] = 0; gz8[6] = 0; gz8[7] = 0;
    __builtin_amdgcn_tensor_load_to_lds(g0, g1, gz4, gz4, gz8, 0);
}
#endif

// ---------------------------------------------------------------------------
// Utility kernels
// ---------------------------------------------------------------------------

// W[K][N] f32 -> Wt[N][K] f16
__global__ void wtrans_kernel(const float* __restrict__ w, _Float16* __restrict__ wt,
                              int K, int N) {
    size_t idx = (size_t)blockIdx.x * blockDim.x + threadIdx.x;
    if (idx >= (size_t)K * N) return;
    int k = (int)(idx / N), n = (int)(idx % N);
    wt[(size_t)n * K + k] = (_Float16)w[idx];
}

__global__ void fconv_kernel(const float* __restrict__ in, _Float16* __restrict__ out,
                             size_t n) {
    size_t idx = (size_t)blockIdx.x * blockDim.x + threadIdx.x;
    if (idx < n) out[idx] = (_Float16)in[idx];
}

// qkv f16 [B*T][3C] -> vt [BH][64][T]  (v at column offset 2C + h*64)
__global__ void vtrans_self_kernel(const _Float16* __restrict__ qkvh,
                                   _Float16* __restrict__ vt) {
    size_t idx = (size_t)blockIdx.x * blockDim.x + threadIdx.x;
    if (idx >= (size_t)32 * 64 * 2048) return;
    int t  = (int)(idx & 2047);
    int d  = (int)((idx >> 11) & 63);
    int bh = (int)(idx >> 17);
    int b = bh >> 4, h = bh & 15;
    vt[idx] = qkvh[((size_t)(b * 2048 + t)) * 3072 + 2048 + h * 64 + d];
}

// v f16 [B*S][C] -> vt [BH][64][S]
__global__ void vtrans_ca_kernel(const _Float16* __restrict__ v,
                                 _Float16* __restrict__ vt) {
    size_t idx = (size_t)blockIdx.x * blockDim.x + threadIdx.x;
    if (idx >= (size_t)32 * 64 * 576) return;
    int s  = (int)(idx % 576);
    int d  = (int)((idx / 576) & 63);
    int bh = (int)(idx / (576 * 64));
    int b = bh >> 4, h = bh & 15;
    vt[idx] = v[((size_t)(b * 576 + s)) * 1024 + h * 64 + d];
}

// LayerNorm over last dim (C), f32 in -> f16 out
__global__ void ln_kernel(const float* __restrict__ x, const float* __restrict__ w,
                          const float* __restrict__ b, _Float16* __restrict__ out,
                          int C_) {
    int row = blockIdx.x;
    const float* xr = x + (size_t)row * C_;
    __shared__ float red[256];
    float s = 0.f;
    for (int c = threadIdx.x; c < C_; c += 256) s += xr[c];
    red[threadIdx.x] = s; __syncthreads();
    for (int o = 128; o > 0; o >>= 1) {
        if (threadIdx.x < o) red[threadIdx.x] += red[threadIdx.x + o];
        __syncthreads();
    }
    float mean = red[0] * (1.0f / C_);
    __syncthreads();
    float v = 0.f;
    for (int c = threadIdx.x; c < C_; c += 256) { float d = xr[c] - mean; v += d * d; }
    red[threadIdx.x] = v; __syncthreads();
    for (int o = 128; o > 0; o >>= 1) {
        if (threadIdx.x < o) red[threadIdx.x] += red[threadIdx.x + o];
        __syncthreads();
    }
    float rstd = rsqrtf(red[0] * (1.0f / C_) + 1e-5f);
    _Float16* orow = out + (size_t)row * C_;
    for (int c = threadIdx.x; c < C_; c += 256)
        orow[c] = (_Float16)((xr[c] - mean) * rstd * w[c] + b[c]);
}

// ---------------------------------------------------------------------------
// Tiled f16 GEMM: out = A[M][K] * Bt[N][K]^T + bias(+resid)(+gelu)
// block 256 (8 waves), tile 128x128, K-step 32, double-buffered LDS.
// Tile staging via Tensor Data Mover (one descriptor per 128x32 tile, issued
// by wave 0, tracked with TENSORcnt) so compute waves issue only DS + WMMA.
// ---------------------------------------------------------------------------
__global__ void gemm16_kernel(const _Float16* __restrict__ A,
                              const _Float16* __restrict__ Bt,
                              const float* __restrict__ bias,
                              const float* __restrict__ resid,
                              float* __restrict__ out32,
                              _Float16* __restrict__ out16,
                              int M, int N, int K, int dogelu) {
    __shared__ __align__(16) _Float16 As[2][128][32];
    __shared__ __align__(16) _Float16 Bs[2][128][32];
    int n0 = blockIdx.x * 128, m0 = blockIdx.y * 128;
    int t = threadIdx.x;

    int w = t >> 5, l = t & 31;
    int lane15 = l & 15, hi = l >> 4, klo = hi * 8, khi = hi * 16;
    int wm = (w & 3) * 32, wn = (w >> 2) * 64;

    v8f acc[2][4];
    #pragma unroll
    for (int i = 0; i < 2; i++)
        #pragma unroll
        for (int j = 0; j < 4; j++) acc[i][j] = vzero8();

    int nk = K >> 5;

#if USE_TDM
    bool wv0 = (w == 0);
    // generic->LDS pointer: low 32 bits are the LDS byte offset
    unsigned ldsA = (unsigned)(uintptr_t)&As[0][0][0];
    unsigned ldsB = (unsigned)(uintptr_t)&Bs[0][0][0];
    const _Float16* gA = A + (size_t)m0 * K;
    const _Float16* gB = Bt + (size_t)n0 * K;
    if (wv0) {
        tdm_load_tile_2d(ldsA, gA, (unsigned)K, (unsigned)M, 32u, 128u, (unsigned)K);
        tdm_load_tile_2d(ldsB, gB, (unsigned)K, (unsigned)N, 32u, 128u, (unsigned)K);
    }
    for (int ki = 0; ki < nk; ki++) {
        int s = ki & 1;
        if (wv0) {
            if (ki + 1 < nk) {
                int kk = (ki + 1) * 32;
                unsigned boff = (unsigned)((s ^ 1) * 128 * 32 * 2);
                tdm_load_tile_2d(ldsA + boff, gA + kk, (unsigned)K, (unsigned)M,
                                 32u, 128u, (unsigned)K);
                tdm_load_tile_2d(ldsB + boff, gB + kk, (unsigned)K, (unsigned)N,
                                 32u, 128u, (unsigned)K);
                __builtin_amdgcn_s_wait_tensorcnt((short)2);  // stage s landed
            } else {
                __builtin_amdgcn_s_wait_tensorcnt((short)0);
            }
        }
        __syncthreads();  // stage s visible to all waves
        v16h a0 = frag_a(&As[s][wm + lane15][0], klo);
        v16h a1 = frag_a(&As[s][wm + 16 + lane15][0], klo);
        #pragma unroll
        for (int j = 0; j < 4; j++) {
            v16h bb = frag_b(&Bs[s][wn + j * 16 + lane15][khi]);
            acc[0][j] = wmma_f16(a0, bb, acc[0][j]);
            acc[1][j] = wmma_f16(a1, bb, acc[1][j]);
        }
        __syncthreads();  // reads of stage s done before TDM overwrites it
    }
#else
    int lr = t >> 1, lp = (t & 1) * 16;
    const _Float16* ga = A  + (size_t)(m0 + lr) * K + lp;
    const _Float16* gb = Bt + (size_t)(n0 + lr) * K + lp;
    auto stage = [&](int s, int kk) {
        *(h8*)&As[s][lr][lp]     = *(const h8*)(ga + kk);
        *(h8*)&As[s][lr][lp + 8] = *(const h8*)(ga + kk + 8);
        *(h8*)&Bs[s][lr][lp]     = *(const h8*)(gb + kk);
        *(h8*)&Bs[s][lr][lp + 8] = *(const h8*)(gb + kk + 8);
    };
    stage(0, 0);
    __syncthreads();
    for (int ki = 0; ki < nk; ki++) {
        int s = ki & 1;
        if (ki + 1 < nk) stage(s ^ 1, (ki + 1) * 32);
        v16h a0 = frag_a(&As[s][wm + lane15][0], klo);
        v16h a1 = frag_a(&As[s][wm + 16 + lane15][0], klo);
        #pragma unroll
        for (int j = 0; j < 4; j++) {
            v16h bb = frag_b(&Bs[s][wn + j * 16 + lane15][khi]);
            acc[0][j] = wmma_f16(a0, bb, acc[0][j]);
            acc[1][j] = wmma_f16(a1, bb, acc[1][j]);
        }
        __syncthreads();
    }
#endif

    #pragma unroll
    for (int i = 0; i < 2; i++)
        #pragma unroll
        for (int j = 0; j < 4; j++)
            #pragma unroll
            for (int r = 0; r < 8; r++) {
                int row = m0 + wm + i * 16 + r + hi * 8;
                int col = n0 + wn + j * 16 + lane15;
                float v = acc[i][j][r] + bias[col];
                if (dogelu) v = gelu_tanh(v);
                size_t idx = (size_t)row * N + col;
                if (resid) v += resid[idx];
                if (out32) out32[idx] = v;
                if (out16) out16[idx] = (_Float16)v;
            }
}

// ---------------------------------------------------------------------------
// Flash causal self-attention. grid (T/64, B*H), block 128 (4 waves),
// each wave owns 16 query rows, independent (no block barriers).
// ---------------------------------------------------------------------------
__global__ void flash_self_kernel(const _Float16* __restrict__ qkvh,
                                  const _Float16* __restrict__ vt,
                                  _Float16* __restrict__ sa) {
    const int TT = 2048;
    int bh = blockIdx.y, b = bh >> 4, h = bh & 15;
    int w = threadIdx.x >> 5, l = threadIdx.x & 31;
    int lane15 = l & 15, hi = l >> 4, klo = hi * 8;
    int qrow0 = blockIdx.x * 64 + w * 16;
    __shared__ __align__(16) _Float16 pst[4][16][32];

    const _Float16* qp = qkvh + ((size_t)(b * TT + qrow0 + lane15)) * 3072 + h * 64;
    v16h qa0 = frag_a(qp,      klo);
    v16h qa1 = frag_a(qp + 32, klo);

    v8f o0 = vzero8(), o1 = vzero8(), o2 = vzero8(), o3 = vzero8();
    float m_run[8], l_run[8];
    #pragma unroll
    for (int r = 0; r < 8; r++) { m_run[r] = -1e30f; l_run[r] = 0.f; }

    const _Float16* kbase = qkvh + ((size_t)b * TT) * 3072 + 1024 + h * 64;
    const _Float16* vbase = vt + ((size_t)bh * 64) * TT;

    int nkt = (qrow0 + 47) >> 5;  // causal: keys up to qrow0+15
    for (int kt = 0; kt < nkt; kt++) {
        int k0 = kt * 32;
        v8f s0 = vzero8(), s1 = vzero8();
        {
            const _Float16* kp0 = kbase + (size_t)(k0 + lane15) * 3072 + hi * 16;
            s0 = wmma_f16(qa0, frag_b(kp0),      s0);
            s0 = wmma_f16(qa1, frag_b(kp0 + 32), s0);
            const _Float16* kp1 = kbase + (size_t)(k0 + 16 + lane15) * 3072 + hi * 16;
            s1 = wmma_f16(qa0, frag_b(kp1),      s1);
            s1 = wmma_f16(qa1, frag_b(kp1 + 32), s1);
        }
        #pragma unroll
        for (int r = 0; r < 8; r++) {
            int qq = qrow0 + r + hi * 8;
            float a = s0[r] * 0.125f; if (k0 + lane15 > qq)       a = -1e30f;
            float c = s1[r] * 0.125f; if (k0 + 16 + lane15 > qq)  c = -1e30f;
            float rm = fmaxf(a, c);
            rm = fmaxf(rm, __shfl_xor(rm, 1, 16));
            rm = fmaxf(rm, __shfl_xor(rm, 2, 16));
            rm = fmaxf(rm, __shfl_xor(rm, 4, 16));
            rm = fmaxf(rm, __shfl_xor(rm, 8, 16));
            float mn = fmaxf(m_run[r], rm);
            float alpha = __expf(m_run[r] - mn);
            m_run[r] = mn;
            float p0 = __expf(a - mn);
            float p1 = __expf(c - mn);
            float ps = p0 + p1;
            ps += __shfl_xor(ps, 1, 16);
            ps += __shfl_xor(ps, 2, 16);
            ps += __shfl_xor(ps, 4, 16);
            ps += __shfl_xor(ps, 8, 16);
            l_run[r] = l_run[r] * alpha + ps;
            o0[r] *= alpha; o1[r] *= alpha; o2[r] *= alpha; o3[r] *= alpha;
            pst[w][r + hi * 8][lane15]      = (_Float16)p0;
            pst[w][r + hi * 8][16 + lane15] = (_Float16)p1;
        }
        // per-wave LDS transpose: ds ops are in-order within a wave, so a
        // dscnt drain is sufficient cross-lane sync (no block barrier needed)
        __asm__ volatile("s_wait_dscnt 0x0" ::: "memory");
        v16h pa = frag_a(&pst[w][lane15][0], klo);
        const _Float16* vp = vbase + (size_t)lane15 * TT + k0 + hi * 16;
        o0 = wmma_f16(pa, frag_b(vp),           o0);
        o1 = wmma_f16(pa, frag_b(vp + 16 * TT), o1);
        o2 = wmma_f16(pa, frag_b(vp + 32 * TT), o2);
        o3 = wmma_f16(pa, frag_b(vp + 48 * TT), o3);
    }

    #pragma unroll
    for (int r = 0; r < 8; r++) {
        int row = qrow0 + r + hi * 8;
        float inv = 1.0f / l_run[r];
        _Float16* op = sa + ((size_t)(b * TT + row)) * 1024 + h * 64 + lane15;
        op[0]  = (_Float16)(o0[r] * inv);
        op[16] = (_Float16)(o1[r] * inv);
        op[32] = (_Float16)(o2[r] * inv);
        op[48] = (_Float16)(o3[r] * inv);
    }
}

// ---------------------------------------------------------------------------
// Cross-attention with exact softmax (S=576). grid (T/16, B*H), block 32.
// Dynamic LDS: scores f32 [16][576] + probs f16 [16][576] = 55296 B.
// Writes normalized probabilities to d_out's attn_prob region.
// ---------------------------------------------------------------------------
__global__ void cross_attn_kernel(const _Float16* __restrict__ q16,
                                  const _Float16* __restrict__ k16,
                                  const _Float16* __restrict__ vt,
                                  _Float16* __restrict__ ca,
                                  float* __restrict__ probs) {
    extern __shared__ __align__(16) char smem[];
    float*    sc = (float*)smem;                      // [16][576]
    _Float16* ps = (_Float16*)(smem + 16 * 576 * 4);  // [16][576]
    int bh = blockIdx.y, b = bh >> 4, h = bh & 15;
    int l = threadIdx.x, lane15 = l & 15, hi = l >> 4, klo = hi * 8;
    int qrow0 = blockIdx.x * 16;

    const _Float16* qp = q16 + ((size_t)(b * 2048 + qrow0 + lane15)) * 1024 + h * 64;
    v16h qa0 = frag_a(qp,      klo);
    v16h qa1 = frag_a(qp + 32, klo);
    const _Float16* kbase = k16 + ((size_t)b * 576) * 1024 + h * 64;

    for (int nt = 0; nt < 36; nt++) {
        v8f s = vzero8();
        const _Float16* kp = kbase + (size_t)(nt * 16 + lane15) * 1024 + hi * 16;
        s = wmma_f16(qa0, frag_b(kp),      s);
        s = wmma_f16(qa1, frag_b(kp + 32), s);
        #pragma unroll
        for (int r = 0; r < 8; r++)
            sc[(r + hi * 8) * 576 + nt * 16 + lane15] = s[r] * 0.125f;
    }
    __asm__ volatile("s_wait_dscnt 0x0" ::: "memory");

    // softmax: lane handles row (l&15), half hi of 576 columns
    {
        int row = lane15, c0 = hi * 288;
        float mx = -1e30f;
        for (int c = 0; c < 288; c++) mx = fmaxf(mx, sc[row * 576 + c0 + c]);
        mx = fmaxf(mx, __shfl_xor(mx, 16, 32));
        float sum = 0.f;
        for (int c = 0; c < 288; c++) {
            float e = __expf(sc[row * 576 + c0 + c] - mx);
            sc[row * 576 + c0 + c] = e;
            sum += e;
        }
        sum += __shfl_xor(sum, 16, 32);
        float inv = 1.0f / sum;
        float* pout = probs + ((size_t)bh * 2048 + qrow0 + row) * 576 + c0;
        for (int c = 0; c < 288; c++) {
            float p = sc[row * 576 + c0 + c] * inv;
            pout[c] = p;
            ps[row * 576 + c0 + c] = (_Float16)p;
        }
    }
    __asm__ volatile("s_wait_dscnt 0x0" ::: "memory");

    v8f o0 = vzero8(), o1 = vzero8(), o2 = vzero8(), o3 = vzero8();
    const _Float16* vbase = vt + (size_t)bh * 64 * 576;
    for (int kt = 0; kt < 18; kt++) {
        v16h pa = frag_a(ps + (size_t)lane15 * 576 + kt * 32, klo);
        const _Float16* vp = vbase + (size_t)lane15 * 576 + kt * 32 + hi * 16;
        o0 = wmma_f16(pa, frag_b(vp),            o0);
        o1 = wmma_f16(pa, frag_b(vp + 16 * 576), o1);
        o2 = wmma_f16(pa, frag_b(vp + 32 * 576), o2);
        o3 = wmma_f16(pa, frag_b(vp + 48 * 576), o3);
    }
    #pragma unroll
    for (int r = 0; r < 8; r++) {
        int row = qrow0 + r + hi * 8;
        _Float16* op = ca + ((size_t)(b * 2048 + row)) * 1024 + h * 64 + lane15;
        op[0]  = (_Float16)o0[r];
        op[16] = (_Float16)o1[r];
        op[32] = (_Float16)o2[r];
        op[48] = (_Float16)o3[r];
    }
}

// ---------------------------------------------------------------------------
// Host orchestration
// ---------------------------------------------------------------------------
extern "C" void kernel_launch(void* const* d_in, const int* in_sizes, int n_in,
                              void* d_out, int out_size, void* d_ws, size_t ws_size,
                              hipStream_t stream) {
    (void)in_sizes; (void)n_in; (void)out_size; (void)ws_size;
    const int Bb = 2, T = 2048, S = 576, C = 1024;
    const int NT = Bb * T;   // 4096 token rows
    const int NS = Bb * S;   // 1152 context rows

    const float* x        = (const float*)d_in[0];
    const float* ctx      = (const float*)d_in[1];
    const float* ln1w     = (const float*)d_in[2];
    const float* ln1b     = (const float*)d_in[3];
    const float* ln2w     = (const float*)d_in[4];
    const float* ln2b     = (const float*)d_in[5];
    const float* ln3w     = (const float*)d_in[6];
    const float* ln3b     = (const float*)d_in[7];
    const float* wqkv     = (const float*)d_in[8];
    const float* bqkv     = (const float*)d_in[9];
    const float* wproj    = (const float*)d_in[10];
    const float* bproj    = (const float*)d_in[11];
    const float* ca_wq    = (const float*)d_in[12];
    const float* ca_bq    = (const float*)d_in[13];
    const float* ca_wk    = (const float*)d_in[14];
    const float* ca_bk    = (const float*)d_in[15];
    const float* ca_wv    = (const float*)d_in[16];
    const float* ca_bv    = (const float*)d_in[17];
    const float* ca_wp    = (const float*)d_in[18];
    const float* ca_bp    = (const float*)d_in[19];
    const float* fc_w     = (const float*)d_in[20];
    const float* fc_b     = (const float*)d_in[21];
    const float* pw_w     = (const float*)d_in[22];
    const float* pw_b     = (const float*)d_in[23];

    float* outx  = (float*)d_out;
    float* probs = outx + (size_t)Bb * T * C;   // attn_prob region [B,H,T,S]

    char* ws = (char*)d_ws;
    size_t off = 0;
    auto alloc = [&](size_t bytes) -> char* {
        char* p = ws + off;
        off += (bytes + 255) & ~(size_t)255;
        return p;
    };
    _Float16* wqkv_t  = (_Float16*)alloc((size_t)C * 3 * C * 2);
    _Float16* wproj_t = (_Float16*)alloc((size_t)C * C * 2);
    _Float16* wq_t    = (_Float16*)alloc((size_t)C * C * 2);
    _Float16* wk_t    = (_Float16*)alloc((size_t)C * C * 2);
    _Float16* wv_t    = (_Float16*)alloc((size_t)C * C * 2);
    _Float16* wcp_t   = (_Float16*)alloc((size_t)C * C * 2);
    _Float16* wfc_t   = (_Float16*)alloc((size_t)C * 4 * C * 2);
    _Float16* wpw_t   = (_Float16*)alloc((size_t)4 * C * C * 2);
    _Float16* hbuf    = (_Float16*)alloc((size_t)NT * C * 2);
    _Float16* ctxh    = (_Float16*)alloc((size_t)NS * C * 2);
    _Float16* qkvh    = (_Float16*)alloc((size_t)NT * 3 * C * 2);
    _Float16* vt_self = (_Float16*)alloc((size_t)32 * 64 * T * 2);
    _Float16* sa16    = (_Float16*)alloc((size_t)NT * C * 2);
    float*    x1      = (float*)   alloc((size_t)NT * C * 4);
    _Float16* qca16   = (_Float16*)alloc((size_t)NT * C * 2);
    _Float16* kca16   = (_Float16*)alloc((size_t)NS * C * 2);
    _Float16* vca16   = (_Float16*)alloc((size_t)NS * C * 2);
    _Float16* vtca    = (_Float16*)alloc((size_t)32 * 64 * S * 2);
    _Float16* ca16    = (_Float16*)alloc((size_t)NT * C * 2);
    float*    x2      = (float*)   alloc((size_t)NT * C * 4);
    _Float16* hfc     = (_Float16*)alloc((size_t)NT * 4 * C * 2);

    auto gblk = [](size_t n) { return (unsigned)((n + 255) / 256); };

    // weight conversion/transpose (f32 [K][N] -> f16 [N][K])
    wtrans_kernel<<<gblk((size_t)C * 3 * C), 256, 0, stream>>>(wqkv,  wqkv_t,  C, 3 * C);
    wtrans_kernel<<<gblk((size_t)C * C),     256, 0, stream>>>(wproj, wproj_t, C, C);
    wtrans_kernel<<<gblk((size_t)C * C),     256, 0, stream>>>(ca_wq, wq_t,    C, C);
    wtrans_kernel<<<gblk((size_t)C * C),     256, 0, stream>>>(ca_wk, wk_t,    C, C);
    wtrans_kernel<<<gblk((size_t)C * C),     256, 0, stream>>>(ca_wv, wv_t,    C, C);
    wtrans_kernel<<<gblk((size_t)C * C),     256, 0, stream>>>(ca_wp, wcp_t,   C, C);
    wtrans_kernel<<<gblk((size_t)C * 4 * C), 256, 0, stream>>>(fc_w,  wfc_t,   C, 4 * C);
    wtrans_kernel<<<gblk((size_t)4 * C * C), 256, 0, stream>>>(pw_w,  wpw_t,   4 * C, C);
    fconv_kernel<<<gblk((size_t)NS * C), 256, 0, stream>>>(ctx, ctxh, (size_t)NS * C);

    // --- causal self-attention ---
    ln_kernel<<<NT, 256, 0, stream>>>(x, ln1w, ln1b, hbuf, C);
    gemm16_kernel<<<dim3(3 * C / 128, NT / 128), 256, 0, stream>>>(
        hbuf, wqkv_t, bqkv, nullptr, nullptr, qkvh, NT, 3 * C, C, 0);
    vtrans_self_kernel<<<gblk((size_t)32 * 64 * T), 256, 0, stream>>>(qkvh, vt_self);
    flash_self_kernel<<<dim3(T / 64, 32), 128, 0, stream>>>(qkvh, vt_self, sa16);
    gemm16_kernel<<<dim3(C / 128, NT / 128), 256, 0, stream>>>(
        sa16, wproj_t, bproj, x, x1, nullptr, NT, C, C, 0);

    // --- cross-attention ---
    ln_kernel<<<NT, 256, 0, stream>>>(x1, ln2w, ln2b, hbuf, C);
    gemm16_kernel<<<dim3(C / 128, NT / 128), 256, 0, stream>>>(
        hbuf, wq_t, ca_bq, nullptr, nullptr, qca16, NT, C, C, 0);
    gemm16_kernel<<<dim3(C / 128, NS / 128), 256, 0, stream>>>(
        ctxh, wk_t, ca_bk, nullptr, nullptr, kca16, NS, C, C, 0);
    gemm16_kernel<<<dim3(C / 128, NS / 128), 256, 0, stream>>>(
        ctxh, wv_t, ca_bv, nullptr, nullptr, vca16, NS, C, C, 0);
    vtrans_ca_kernel<<<gblk((size_t)32 * 64 * S), 256, 0, stream>>>(vca16, vtca);
    cross_attn_kernel<<<dim3(T / 16, 32), 32, 55296, stream>>>(
        qca16, kca16, vtca, ca16, probs);
    gemm16_kernel<<<dim3(C / 128, NT / 128), 256, 0, stream>>>(
        ca16, wcp_t, ca_bp, x1, x2, nullptr, NT, C, C, 0);

    // --- GELU MLP ---
    ln_kernel<<<NT, 256, 0, stream>>>(x2, ln3w, ln3b, hbuf, C);
    gemm16_kernel<<<dim3(4 * C / 128, NT / 128), 256, 0, stream>>>(
        hbuf, wfc_t, fc_b, nullptr, nullptr, hfc, NT, 4 * C, C, 1);
    gemm16_kernel<<<dim3(C / 128, NT / 128), 256, 0, stream>>>(
        hfc, wpw_t, pw_b, x2, outx, nullptr, NT, C, 4 * C, 0);
}
